// MACNetwork_33500744909515
// MI455X (gfx1250) — matrix operand order repaired and checked by reading, not compiled, not run on hardware
//
#include <hip/hip_runtime.h>
#include <math.h>

typedef __bf16 bf16;
typedef __attribute__((ext_vector_type(16))) __bf16 v16bf;
typedef __attribute__((ext_vector_type(8))) __bf16 v8bf;
typedef __attribute__((ext_vector_type(8))) float v8f;

#define BM 64
#define BN 128
#define BK 32
#define GT 128        // 4 waves of 32; wave grid 2x2, wave tile 32x64
#define SA_STR 40     // padded LDS row stride (elements): 80B/row, 16B aligned
#define SB_STR 40
#define SA_BUF (BM * SA_STR)
#define SB_BUF (BN * SB_STR)

__device__ __forceinline__ float act_elu(float x) { return x > 0.f ? x : (__expf(x) - 1.f); }

#define WMMA_BF16(a, b, c) \
  __builtin_amdgcn_wmma_f32_16x16x32_bf16(false, (a), false, (b), (short)0, (c), false, false)

// ---------------- staging helpers ----------------
// sA: [BM][SA_STR] row-major (m, k).  sBt: [BN][SB_STR] TRANSPOSED (n, k).

// A tile via CDNA5 async DMA: 16 contiguous bf16 (32B) per thread, global -> LDS,
// no VGPR round-trip; tracked by ASYNCcnt. INST_OFFSET applies to both addresses.
__device__ __forceinline__ void async_stage_a(const bf16* __restrict__ A, int lda,
                                              int m0, int k0, bf16* sA) {
  int idx = threadIdx.x * 16;
  int ra = idx >> 5, ca = idx & 31;
  const bf16* gp = A + (long)(m0 + ra) * lda + k0 + ca;
  bf16* lp = &sA[ra * SA_STR + ca];
  unsigned loff = (unsigned)(unsigned long long)(__attribute__((address_space(3))) bf16*)lp;
  asm volatile(
      "global_load_async_to_lds_b128 %0, %1, off\n\t"
      "global_load_async_to_lds_b128 %0, %1, off offset:16"
      :
      : "v"(loff), "v"(gp)
      : "memory");
}

__device__ __forceinline__ void gload_b(const bf16* __restrict__ B, int ldb,
                                        int n0, int k0, v8bf* br) {
  int idx = threadIdx.x * 32;
  int rb = idx >> 7, cb = idx & 127;
  const v8bf* bp = (const v8bf*)(B + (long)(k0 + rb) * ldb + n0 + cb);
  br[0] = bp[0]; br[1] = bp[1]; br[2] = bp[2]; br[3] = bp[3];
}

__device__ __forceinline__ void store_b(bf16* sBt, const v8bf* br) {
  int idx = threadIdx.x * 32;
  int rb = idx >> 7, cb = idx & 127;
#pragma unroll
  for (int g = 0; g < 4; ++g)
#pragma unroll
    for (int e = 0; e < 8; ++e) sBt[(cb + g * 8 + e) * SB_STR + rb] = br[g][e];
}

__device__ void stage_slow(const bf16* __restrict__ A, const bf16* __restrict__ B,
                           int M, int N, int K, int lda, int ldb,
                           int m0, int n0, int k0, bf16* sA, bf16* sBt) {
  {
    int idx = threadIdx.x * 16;
    int ra = idx >> 5, ca = idx & 31;
    int gm = m0 + ra;
    const bf16* ap = A + (long)gm * lda + k0;
#pragma unroll 1
    for (int e = 0; e < 16; ++e) {
      int c = ca + e;
      bf16 v = (bf16)0.f;
      if (gm < M && (k0 + c) < K) v = ap[c];
      sA[ra * SA_STR + c] = v;
    }
  }
  {
    int idx = threadIdx.x * 32;
    int rb = idx >> 7, cb = idx & 127;
    int gk = k0 + rb;
    const bf16* bp = B + (long)gk * ldb + n0;
#pragma unroll 1
    for (int e = 0; e < 32; ++e) {
      int c = cb + e;
      bf16 v = (bf16)0.f;
      if (gk < K && (n0 + c) < N) v = bp[c];
      sBt[c * SB_STR + rb] = v;
    }
  }
}

// A fragment: 16x32 bf16 per ISA table; lane half selects K-block. Two b128 LDS reads.
__device__ __forceinline__ v16bf frag_a(const bf16* sA, int mbase, int lane) {
  int m = mbase + (lane & 15);
  int half = (lane >> 4) & 1;
  v8bf lo = *(const v8bf*)&sA[m * SA_STR + half * 8];
  v8bf hi = *(const v8bf*)&sA[m * SA_STR + 16 + half * 8];
  return __builtin_shufflevector(lo, hi, 0, 1, 2, 3, 4, 5, 6, 7, 8, 9, 10, 11, 12, 13, 14, 15);
}

// B fragment: per ISA table, lane half selects K-block (0-7,16-23 / 8-15,24-31).
__device__ __forceinline__ v16bf frag_b(const bf16* sBt, int nbase, int lane) {
  int n = nbase + (lane & 15);
  int half = (lane >> 4) & 1;
  v8bf lo = *(const v8bf*)&sBt[n * SB_STR + half * 8];
  v8bf hi = *(const v8bf*)&sBt[n * SB_STR + 16 + half * 8];
  return __builtin_shufflevector(lo, hi, 0, 1, 2, 3, 4, 5, 6, 7, 8, 9, 10, 11, 12, 13, 14, 15);
}

// one k-step of compute for a 32x64 wave tile: 8 WMMA from 2 A-frags + 4 B-frags
__device__ __forceinline__ void compute_tile(const bf16* shA, const bf16* shB,
                                             int wm, int wn, int lane, v8f (*acc)[4]) {
  v16bf a0 = frag_a(shA, wm * 32, lane);
  v16bf a1 = frag_a(shA, wm * 32 + 16, lane);
  v16bf b0 = frag_b(shB, wn * 64, lane);
  v16bf b1 = frag_b(shB, wn * 64 + 16, lane);
  v16bf b2 = frag_b(shB, wn * 64 + 32, lane);
  v16bf b3 = frag_b(shB, wn * 64 + 48, lane);
  acc[0][0] = WMMA_BF16(a0, b0, acc[0][0]);
  acc[0][1] = WMMA_BF16(a0, b1, acc[0][1]);
  acc[0][2] = WMMA_BF16(a0, b2, acc[0][2]);
  acc[0][3] = WMMA_BF16(a0, b3, acc[0][3]);
  acc[1][0] = WMMA_BF16(a1, b0, acc[1][0]);
  acc[1][1] = WMMA_BF16(a1, b1, acc[1][1]);
  acc[1][2] = WMMA_BF16(a1, b2, acc[1][2]);
  acc[1][3] = WMMA_BF16(a1, b3, acc[1][3]);
}

// Shared GEMM mainloop. Fast path: single barrier per k-step, double-buffered LDS,
// async-DMA A prefetch + register-prefetched B overlapping the 8 WMMAs.
__device__ __forceinline__ void gemm_core(const bf16* __restrict__ Ab, const bf16* __restrict__ Bb,
                                          int M, int N, int K, int lda, int ldb,
                                          int m0, int n0, int lane, int wm, int wn,
                                          bf16* shA, bf16* shB, v8f (*acc)[4]) {
  bool fast = (m0 + BM <= M) && (n0 + BN <= N) && ((K & (BK - 1)) == 0) &&
              ((lda & 7) == 0) && ((ldb & 7) == 0);
  if (fast) {
    v8bf br[4];
    async_stage_a(Ab, lda, m0, 0, shA);
    gload_b(Bb, ldb, n0, 0, br);
    int cur = 0;
    for (int k0 = 0; k0 < K; k0 += BK) {
      store_b(shB + cur * SB_BUF, br);
      asm volatile("s_wait_asynccnt 0" ::: "memory");
      __syncthreads();  // all waves' async-A landed + B stores visible
      int kn = k0 + BK;
      if (kn < K) {
        async_stage_a(Ab, lda, m0, kn, shA + (1 - cur) * SA_BUF);
        gload_b(Bb, ldb, n0, kn, br);
      }
      compute_tile(shA + cur * SA_BUF, shB + cur * SB_BUF, wm, wn, lane, acc);
      cur ^= 1;
    }
  } else {
    for (int k0 = 0; k0 < K; k0 += BK) {
      __syncthreads();
      stage_slow(Ab, Bb, M, N, K, lda, ldb, m0, n0, k0, shA, shB);
      __syncthreads();
      compute_tile(shA, shB, wm, wn, lane, acc);
    }
  }
}

// ---------------- generic bf16 WMMA GEMM ----------------
// C = act(A*B + bias + Cadd), batched via blockIdx.z with element strides.
// ACT: 0=none 1=ELU 2=tanh.  OBF: bf16 output, else f32.
template <int ACT, bool BIAS, bool CADD, bool OBF>
__global__ void gemm_bf16(const bf16* __restrict__ A, const bf16* __restrict__ B,
                          const float* __restrict__ bias, const float* __restrict__ Cadd,
                          void* __restrict__ Cout, int M, int N, int K,
                          int lda, int ldb, int ldc,
                          long strA, long strB, long strAdd, long strC) {
  __shared__ bf16 shA[2 * SA_BUF];
  __shared__ bf16 shB[2 * SB_BUF];
  int z = blockIdx.z;
  const bf16* Ab = A + (long)z * strA;
  const bf16* Bb = B + (long)z * strB;
  const float* Db = CADD ? (Cadd + (long)z * strAdd) : nullptr;
  int m0 = blockIdx.y * BM, n0 = blockIdx.x * BN;
  int lane = threadIdx.x & 31, wid = threadIdx.x >> 5;
  int wm = wid >> 1, wn = wid & 1;
  v8f acc[2][4] = {};
  gemm_core(Ab, Bb, M, N, K, lda, ldb, m0, n0, lane, wm, wn, shA, shB, acc);
  int ln = lane & 15, lm = (lane >> 4) * 8;
#pragma unroll
  for (int mi = 0; mi < 2; ++mi)
#pragma unroll
    for (int ni = 0; ni < 4; ++ni)
#pragma unroll
      for (int v = 0; v < 8; ++v) {
        int m = m0 + wm * 32 + mi * 16 + lm + v;
        int n = n0 + wn * 64 + ni * 16 + ln;
        if (m < M && n < N) {
          float x = acc[mi][ni][v];
          if (BIAS) x += bias[n];
          if (CADD) x += Db[(long)m * ldc + n];
          if (ACT == 1) x = act_elu(x);
          else if (ACT == 2) x = tanhf(x);
          if (OBF)
            ((bf16*)Cout)[(long)z * strC + (long)m * ldc + n] = (bf16)x;
          else
            ((float*)Cout)[(long)z * strC + (long)m * ldc + n] = x;
        }
      }
}

// ---------------- read-unit GEMM2 with fused attention-logit reduction ----------------
// acc = inter1 @ Wcat2 ; t = elu((acc+bcat2)*c[b]) ; rl[b,m] += sum_n t*rw[n]
__global__ void gemm_read_attn(const bf16* __restrict__ A, const bf16* __restrict__ B,
                               const float* __restrict__ bcat2, const float* __restrict__ cvec,
                               const float* __restrict__ rw, float* __restrict__ rl,
                               int M, int N, int K, int lda, int ldb, long strA) {
  __shared__ bf16 shA[2 * SA_BUF];
  __shared__ bf16 shB[2 * SB_BUF];
  int z = blockIdx.z;
  const bf16* Ab = A + (long)z * strA;
  int m0 = blockIdx.y * BM, n0 = blockIdx.x * BN;
  int lane = threadIdx.x & 31, wid = threadIdx.x >> 5;
  int wm = wid >> 1, wn = wid & 1;
  v8f acc[2][4] = {};
  gemm_core(Ab, B, M, N, K, lda, ldb, m0, n0, lane, wm, wn, shA, shB, acc);
  int ln = lane & 15, lm = (lane >> 4) * 8;
  const float* cb = cvec + (long)z * 512;
#pragma unroll
  for (int mi = 0; mi < 2; ++mi)
#pragma unroll
    for (int v = 0; v < 8; ++v) {
      float s = 0.f;
#pragma unroll
      for (int ni = 0; ni < 4; ++ni) {
        int n = n0 + wn * 64 + ni * 16 + ln;
        float x = acc[mi][ni][v] + bcat2[n];
        x *= cb[n];
        x = act_elu(x);
        s += x * rw[n];
      }
#pragma unroll
      for (int off = 1; off < 16; off <<= 1) s += __shfl_xor(s, off, 32);
      int m = m0 + wm * 32 + mi * 16 + lm + v;
      if (ln == 0 && m < M) atomicAdd(&rl[(long)z * M + m], s);
    }
}

// ---------------- small helper kernels ----------------
__global__ void k_cvt_bf16(const float* __restrict__ in, bf16* __restrict__ out, long n) {
  long i = (long)blockIdx.x * blockDim.x + threadIdx.x;
  if (i < n) out[i] = (bf16)in[i];
}

__global__ void k_bcast(const float* __restrict__ src, float* __restrict__ dst, long rows, int cols) {
  long i = (long)blockIdx.x * blockDim.x + threadIdx.x;
  if (i < rows * cols) dst[i] = src[i % cols];
}

__global__ void k_concat2_bf16(const float* __restrict__ a, const float* __restrict__ b,
                               bf16* __restrict__ out, int rows, int cols) {
  long i = (long)blockIdx.x * blockDim.x + threadIdx.x;
  long tot = (long)rows * 2 * cols;
  if (i >= tot) return;
  int r = (int)(i / (2 * cols));
  int c = (int)(i % (2 * cols));
  float v = (c < cols) ? a[(long)r * cols + c] : b[(long)r * cols + (c - cols)];
  out[i] = (bf16)v;
}

__global__ void k_colscale(const float* __restrict__ mat, const float* __restrict__ vec,
                           float* __restrict__ out, long rows, int cols) {
  long i = (long)blockIdx.x * blockDim.x + threadIdx.x;
  if (i >= rows * cols) return;
  out[i] = mat[i] * vec[i % cols];
}

__global__ void k_weff(const float* __restrict__ mp, const float* __restrict__ WcatTop,
                       bf16* __restrict__ out) {
  long i = (long)blockIdx.x * blockDim.x + threadIdx.x;
  if (i >= (long)32 * 512 * 512) return;
  int b = (int)(i >> 18);
  int r = (int)((i >> 9) & 511);
  int c = (int)(i & 511);
  out[i] = (bf16)(mp[b * 512 + r] * WcatTop[r * 512 + c]);
}

__global__ void k_fill(float* __restrict__ out, const float* __restrict__ sc, long n) {
  long i = (long)blockIdx.x * blockDim.x + threadIdx.x;
  if (i < n) out[i] = sc[0];
}

__global__ void k_ctrl_logits(const float* __restrict__ q, const float* __restrict__ cw,
                              const float* __restrict__ cb, float* __restrict__ cl) {
  int i = blockIdx.x * blockDim.x + threadIdx.x;
  if (i >= 32 * 80) return;
  int b = i / 80, l = i % 80;
  const float* qp = q + b * 512;
  const float* cp = cw + l * 512;
  float s = cb[0];
  for (int d = 0; d < 512; ++d) s += qp[d] * cp[d];
  cl[i] = s;
}

__global__ void k_cvec(const float* __restrict__ attn, const float* __restrict__ concepts,
                       float* __restrict__ c) {
  int i = blockIdx.x * blockDim.x + threadIdx.x;
  if (i >= 32 * 512) return;
  int b = i >> 9, d = i & 511;
  float s = 0.f;
  for (int l = 0; l < 80; ++l) s += attn[b * 80 + l] * concepts[l * 512 + d];
  c[i] = s;
}

__global__ void k_info(const float* __restrict__ attn, const bf16* __restrict__ know,
                       float* __restrict__ info) {
  int i = blockIdx.x * blockDim.x + threadIdx.x;
  if (i >= 32 * 512) return;
  int b = i >> 9, d = i & 511;
  const bf16* kp = know + (long)b * 1568 * 512 + d;
  const float* ap = attn + b * 1568;
  float s = 0.f;
  for (int k = 0; k < 1568; ++k) s += ap[k] * (float)kp[(long)k * 512];
  info[i] = s;
}

__global__ void k_logits(const float* __restrict__ mpo, const float* __restrict__ lw,
                         const float* __restrict__ ob, float* __restrict__ out) {
  int i = blockIdx.x * blockDim.x + threadIdx.x;
  if (i >= 32 * 1000) return;
  int b = i / 1000, a = i % 1000;
  const float* mpp = mpo + b * 300;
  const float* lp = lw + a * 300;
  float s = ob[0];
  for (int j = 0; j < 300; ++j) s += mpp[j] * lp[j];
  out[i] = s;
}

__global__ void k_softmax_rows(const float* __restrict__ in, float* __restrict__ out, int cols) {
  int row = blockIdx.x;
  const float* x = in + (long)row * cols;
  float* y = out + (long)row * cols;
  __shared__ float red[256];
  float mx = -INFINITY;
  for (int j = threadIdx.x; j < cols; j += blockDim.x) mx = fmaxf(mx, x[j]);
  red[threadIdx.x] = mx;
  __syncthreads();
  for (int s = 128; s > 0; s >>= 1) {
    if (threadIdx.x < s) red[threadIdx.x] = fmaxf(red[threadIdx.x], red[threadIdx.x + s]);
    __syncthreads();
  }
  mx = red[0];
  __syncthreads();
  float sum = 0.f;
  for (int j = threadIdx.x; j < cols; j += blockDim.x) sum += __expf(x[j] - mx);
  red[threadIdx.x] = sum;
  __syncthreads();
  for (int s = 128; s > 0; s >>= 1) {
    if (threadIdx.x < s) red[threadIdx.x] += red[threadIdx.x + s];
    __syncthreads();
  }
  float inv = 1.f / red[0];
  for (int j = threadIdx.x; j < cols; j += blockDim.x) y[j] = __expf(x[j] - mx) * inv;
}

// im2col for 3x3x3 SAME conv on 8x14x14; dst[m, kidx*Cin+ci], kidx = kz*9+ky*3+kx.
// layoutSC=0: src is [Cin, 1568]; layoutSC=1: src is [1568, Cin]
__global__ void k_im2col(const float* __restrict__ src, bf16* __restrict__ dst, int Cin, int layoutSC) {
  long i = (long)blockIdx.x * blockDim.x + threadIdx.x;
  long Kc = (long)27 * Cin;
  if (i >= 1568 * Kc) return;
  int m = (int)(i / Kc);
  int r = (int)(i % Kc);
  int kidx = r / Cin, ci = r % Cin;
  int z = m / 196, y = (m / 14) % 14, x = m % 14;
  int zz = z + kidx / 9 - 1, yy = y + (kidx / 3) % 3 - 1, xx = x + kidx % 3 - 1;
  float v = 0.f;
  if ((unsigned)zz < 8u && (unsigned)yy < 14u && (unsigned)xx < 14u) {
    int s = (zz * 14 + yy) * 14 + xx;
    v = layoutSC ? src[(long)s * Cin + ci] : src[(long)ci * 1568 + s];
  }
  dst[i] = (bf16)v;
}

// conv weight [Cout,Cin,3,3,3] -> bf16 [27*Cin, Cout] with row index kidx*Cin+ci
__global__ void k_wreshape(const float* __restrict__ w, bf16* __restrict__ out, int Cout, int Cin) {
  long i = (long)blockIdx.x * blockDim.x + threadIdx.x;
  long tot = (long)Cout * Cin * 27;
  if (i >= tot) return;
  int co = (int)(i / ((long)Cin * 27));
  int rem = (int)(i % ((long)Cin * 27));
  int ci = rem / 27, kidx = rem % 27;
  out[((long)kidx * Cin + ci) * Cout + co] = (bf16)w[i];
}

// ---------------- host ----------------
static inline long cdiv(long a, long b) { return (a + b - 1) / b; }

extern "C" void kernel_launch(void* const* d_in, const int* in_sizes, int n_in,
                              void* d_out, int out_size, void* d_ws, size_t ws_size,
                              hipStream_t stream) {
  (void)in_sizes; (void)n_in; (void)out_size; (void)ws_size;
  const float* image = (const float*)d_in[0];
  const float* concepts = (const float*)d_in[1];
  const float* conv1_w = (const float*)d_in[2];
  const float* conv1_b = (const float*)d_in[3];
  const float* conv2_w = (const float*)d_in[4];
  const float* conv2_b = (const float*)d_in[5];
  const float* Wci = (const float*)d_in[6];
  const float* bci = (const float*)d_in[7];
  const float* Wu = (const float*)d_in[8];
  const float* bu = (const float*)d_in[9];
  const float* ctrl_attn_w = (const float*)d_in[10];
  const float* ctrl_attn_b = (const float*)d_in[11];
  const float* Wk = (const float*)d_in[12];
  const float* bk = (const float*)d_in[13];
  const float* Wm = (const float*)d_in[14];
  const float* bm = (const float*)d_in[15];
  const float* Wcat = (const float*)d_in[16];
  const float* bcat = (const float*)d_in[17];
  const float* Wcat2 = (const float*)d_in[18];
  const float* bcat2 = (const float*)d_in[19];
  const float* read_attn_w = (const float*)d_in[20];
  const float* read_attn_b = (const float*)d_in[21];
  const float* Wwrite = (const float*)d_in[22];
  const float* bwrite = (const float*)d_in[23];
  const float* init_control = (const float*)d_in[24];
  const float* init_memory = (const float*)d_in[25];
  const float* Wproj = (const float*)d_in[26];
  const float* bproj = (const float*)d_in[27];
  const float* out_attn_w = (const float*)d_in[28];
  const float* out_attn_b = (const float*)d_in[29];
  const float* labels = (const float*)d_in[30];

  const int B = 32, Dm = 512, KS = 1568;
  const int K1 = 256 * 27, K2 = 512 * 27;
  const long SKD = (long)KS * Dm;  // 802816

  char* p = (char*)d_ws;
  auto alloc = [&](size_t bytes) -> void* {
    void* r = (void*)p;
    p += (bytes + 255) & ~(size_t)255;
    return r;
  };
  bf16* Wci_b = (bf16*)alloc((size_t)1024 * 512 * 2);
  bf16* Wu_b = (bf16*)alloc((size_t)12 * 512 * 512 * 2);
  bf16* Wk_b = (bf16*)alloc((size_t)512 * 512 * 2);
  bf16* Wm_b = (bf16*)alloc((size_t)512 * 512 * 2);
  bf16* WcatBot_b = (bf16*)alloc((size_t)512 * 512 * 2);
  bf16* Wcat2_b = (bf16*)alloc((size_t)512 * 512 * 2);
  bf16* Wwrite_b = (bf16*)alloc((size_t)1024 * 512 * 2);
  bf16* Wproj_b = (bf16*)alloc((size_t)512 * 300 * 2);
  bf16* W1g = (bf16*)alloc((size_t)K1 * 512 * 2);
  bf16* W2g = (bf16*)alloc((size_t)K2 * 512 * 2);
  float* cw = (float*)alloc((size_t)80 * 512 * 4);
  float* lw = (float*)alloc((size_t)1000 * 300 * 4);
  bf16* imbuf = (bf16*)alloc((size_t)KS * K2 * 2);
  float* x1tmp = (float*)alloc((size_t)KS * 512 * 4);
  bf16* know = (bf16*)alloc((size_t)B * SKD * 2);
  bf16* kproj = (bf16*)alloc((size_t)B * SKD * 2);
  float* kpB = (float*)alloc((size_t)B * SKD * 4);
  bf16* inter1 = (bf16*)alloc((size_t)B * SKD * 2);
  bf16* weff = (bf16*)alloc((size_t)B * 512 * 512 * 2);
  float* cvec = (float*)alloc((size_t)B * 512 * 4);
  float* mvec = (float*)alloc((size_t)B * 512 * 4);
  bf16* mbf = (bf16*)alloc((size_t)B * 512 * 2);
  bf16* catbuf = (bf16*)alloc((size_t)B * 1024 * 2);
  bf16* q1b = (bf16*)alloc((size_t)B * 512 * 2);
  float* q2 = (float*)alloc((size_t)B * 512 * 4);
  float* cl = (float*)alloc((size_t)B * 80 * 4);
  float* mp = (float*)alloc((size_t)B * 512 * 4);
  float* rl = (float*)alloc((size_t)B * KS * 4);
  float* info = (float*)alloc((size_t)B * 512 * 4);
  float* mpo = (float*)alloc((size_t)B * 300 * 4);
  float* logits = (float*)alloc((size_t)B * 1000 * 4);

  dim3 tb(256);
#define G1(n) dim3((unsigned)cdiv((long)(n), 256))

  // ---- weight prep ----
  k_cvt_bf16<<<G1(1024 * 512), tb, 0, stream>>>(Wci, Wci_b, 1024 * 512);
  k_cvt_bf16<<<G1(12 * 512 * 512), tb, 0, stream>>>(Wu, Wu_b, (long)12 * 512 * 512);
  k_cvt_bf16<<<G1(512 * 512), tb, 0, stream>>>(Wk, Wk_b, 512 * 512);
  k_cvt_bf16<<<G1(512 * 512), tb, 0, stream>>>(Wm, Wm_b, 512 * 512);
  k_cvt_bf16<<<G1(512 * 512), tb, 0, stream>>>(Wcat + 512 * 512, WcatBot_b, 512 * 512);
  k_cvt_bf16<<<G1(512 * 512), tb, 0, stream>>>(Wcat2, Wcat2_b, 512 * 512);
  k_cvt_bf16<<<G1(1024 * 512), tb, 0, stream>>>(Wwrite, Wwrite_b, 1024 * 512);
  k_cvt_bf16<<<G1(512 * 300), tb, 0, stream>>>(Wproj, Wproj_b, 512 * 300);
  k_wreshape<<<G1((long)512 * 256 * 27), tb, 0, stream>>>(conv1_w, W1g, 512, 256);
  k_wreshape<<<G1((long)512 * 512 * 27), tb, 0, stream>>>(conv2_w, W2g, 512, 512);
  k_colscale<<<G1(80 * 512), tb, 0, stream>>>(concepts, ctrl_attn_w, cw, 80, 512);
  k_colscale<<<G1(1000 * 300), tb, 0, stream>>>(labels, out_attn_w, lw, 1000, 300);

  // ---- state init ----
  k_bcast<<<G1(B * 512), tb, 0, stream>>>(init_control, cvec, B, 512);
  k_bcast<<<G1(B * 512), tb, 0, stream>>>(init_memory, mvec, B, 512);

  // ---- conv stem (per-batch implicit GEMM via im2col) ----
  for (int b = 0; b < B; ++b) {
    k_im2col<<<G1((long)KS * K1), tb, 0, stream>>>(image + (long)b * 256 * KS, imbuf, 256, 0);
    {
      dim3 g(512 / BN, (unsigned)cdiv(KS, BM), 1);
      gemm_bf16<1, true, false, false><<<g, dim3(GT), 0, stream>>>(
          imbuf, W1g, conv1_b, nullptr, x1tmp, KS, 512, K1, K1, 512, 512, 0, 0, 0, 0);
    }
    k_im2col<<<G1((long)KS * K2), tb, 0, stream>>>(x1tmp, imbuf, 512, 1);
    {
      dim3 g(512 / BN, (unsigned)cdiv(KS, BM), 1);
      gemm_bf16<1, true, false, true><<<g, dim3(GT), 0, stream>>>(
          imbuf, W2g, conv2_b, nullptr, know + (long)b * SKD, KS, 512, K2, K2, 512, 512, 0, 0, 0, 0);
    }
  }

  // ---- step-invariant projections ----
  {
    dim3 g(512 / BN, (unsigned)cdiv(KS, BM), B);
    gemm_bf16<0, true, false, true><<<g, dim3(GT), 0, stream>>>(
        know, Wk_b, bk, nullptr, kproj, KS, 512, 512, 512, 512, 512, SKD, 0, 0, SKD);
    gemm_bf16<0, true, false, false><<<g, dim3(GT), 0, stream>>>(
        kproj, WcatBot_b, bcat, nullptr, kpB, KS, 512, 512, 512, 512, 512, SKD, 0, SKD, SKD);
  }

  // ---- 12 MAC steps ----
  for (int s = 0; s < 12; ++s) {
    // ControlUnit
    k_concat2_bf16<<<G1(B * 1024), tb, 0, stream>>>(cvec, mvec, catbuf, B, 512);
    {
      dim3 g(512 / BN, 1, 1);
      gemm_bf16<2, true, false, true><<<g, dim3(GT), 0, stream>>>(
          catbuf, Wci_b, bci, nullptr, q1b, B, 512, 1024, 1024, 512, 512, 0, 0, 0, 0);
      gemm_bf16<0, true, false, false><<<g, dim3(GT), 0, stream>>>(
          q1b, Wu_b + (long)s * 512 * 512, bu + s * 512, nullptr, q2,
          B, 512, 512, 512, 512, 512, 0, 0, 0, 0);
    }
    k_ctrl_logits<<<G1(B * 80), tb, 0, stream>>>(q2, cw, ctrl_attn_b, cl);
    k_softmax_rows<<<dim3(B), tb, 0, stream>>>(cl, cl, 80);
    k_cvec<<<G1(B * 512), tb, 0, stream>>>(cl, concepts, cvec);

    // ReadUnit
    k_cvt_bf16<<<G1(B * 512), tb, 0, stream>>>(mvec, mbf, B * 512);
    {
      dim3 g(512 / BN, 1, 1);
      gemm_bf16<0, true, false, false><<<g, dim3(GT), 0, stream>>>(
          mbf, Wm_b, bm, nullptr, mp, B, 512, 512, 512, 512, 512, 0, 0, 0, 0);
    }
    k_weff<<<G1((long)B * 512 * 512), tb, 0, stream>>>(mp, Wcat, weff);
    {
      dim3 g(512 / BN, (unsigned)cdiv(KS, BM), B);
      gemm_bf16<1, false, true, true><<<g, dim3(GT), 0, stream>>>(
          kproj, weff, nullptr, kpB, inter1, KS, 512, 512, 512, 512, 512,
          SKD, (long)512 * 512, SKD, SKD);
    }
    k_fill<<<G1(B * KS), tb, 0, stream>>>(rl, read_attn_b, (long)B * KS);
    {
      dim3 g(512 / BN, (unsigned)cdiv(KS, BM), B);
      gemm_read_attn<<<g, dim3(GT), 0, stream>>>(
          inter1, Wcat2_b, bcat2, cvec, read_attn_w, rl, KS, 512, 512, 512, 512, SKD);
    }
    k_softmax_rows<<<dim3(B), tb, 0, stream>>>(rl, rl, KS);
    k_info<<<G1(B * 512), tb, 0, stream>>>(rl, know, info);

    // WriteUnit
    k_concat2_bf16<<<G1(B * 1024), tb, 0, stream>>>(mvec, info, catbuf, B, 512);
    {
      dim3 g(512 / BN, 1, 1);
      gemm_bf16<0, true, false, false><<<g, dim3(GT), 0, stream>>>(
          catbuf, Wwrite_b, bwrite, nullptr, mvec, B, 512, 1024, 1024, 512, 512, 0, 0, 0, 0);
    }
  }

  // ---- OutputUnit ----
  k_cvt_bf16<<<G1(B * 512), tb, 0, stream>>>(mvec, mbf, B * 512);
  {
    dim3 g((unsigned)cdiv(300, BN), 1, 1);
    gemm_bf16<0, true, false, false><<<g, dim3(GT), 0, stream>>>(
        mbf, Wproj_b, bproj, nullptr, mpo, B, 300, 512, 512, 300, 300, 0, 0, 0, 0);
  }
  k_logits<<<G1(B * 1000), tb, 0, stream>>>(mpo, lw, out_attn_b, logits);
  k_softmax_rows<<<dim3(B), tb, 0, stream>>>(logits, (float*)d_out, 1000);
#undef G1
}
